// GBSTLayer_65910568124606
// MI455X (gfx1250) — matrix operand order repaired and checked by reading, not compiled, not run on hardware
//
#include <hip/hip_runtime.h>
#include <hip/hip_bf16.h>

// ---------------------------------------------------------------------------
// GBST layer for MI455X (gfx1250, wave32, WMMA + async-to-LDS staging).
// Pipeline:
//   1. pack_w   : conv_w [E][E][K] f32  -> Wp[k][eo][ei] bf16
//   2. pack_x   : x [B][E][L] f32       -> Xt[b][l][e] bf16 (LDS transpose)
//   3. conv_wmma: y^T[b][l][eo] = sum_k sum_ei Xt[b][l+k][ei]*Wp[k][eo][ei] + bias
//                 v_wmma_f32_16x16x32_bf16, operands staged in LDS via
//                 global_load_async_to_lds_b128 (double buffered, ASYNCcnt)
//   4. score_k  : s[b][l] = dot(yT[b][l][:], score_w)   (wave reduction)
//   5. weights_k: A[b][l][w] = softmax_w(window-mean of s)   (w = 1..4)
//   6. combine_k: out[b][e][n] = 0.5 * sum_{t=0,1} sum_w A[2n+t][w]*pooled_w(2n+t, e)
// ---------------------------------------------------------------------------

#define B_    8
#define E_    512
#define L_    4096
#define KW    5
#define L2_   4092          // L - K + 1, divisible by 1,2,3,4
#define XROWS 4104          // Xt rows: 4096 real + 8 zero pad (tap overreach)
#define ND    2048          // output length after final stride-2 pool

typedef __bf16 bf16t;
typedef bf16t v16bf __attribute__((ext_vector_type(16)));
typedef float v8f   __attribute__((ext_vector_type(8)));

union Frag { v16bf v; uint4 q[2]; };

__device__ __forceinline__ unsigned short f2bf(float f) {
    union { float f; unsigned u; } c; c.f = f;
    unsigned u = c.u;
    unsigned r = ((u >> 16) & 1u) + 0x7FFFu;   // round-to-nearest-even
    return (unsigned short)((u + r) >> 16);
}

// --- CDNA5 async copy: global -> LDS, 16 bytes per lane, ASYNCcnt tracked ---
__device__ __forceinline__ void async_copy16(unsigned lds_off,
                                             const unsigned short* g) {
    asm volatile("global_load_async_to_lds_b128 %0, %1, off"
                 :: "v"(lds_off), "v"((unsigned long long)(size_t)g)
                 : "memory");
}

template <int N>
__device__ __forceinline__ void wait_async() {
    asm volatile("s_wait_asynccnt %0" :: "n"(N) : "memory");
}

// ---------------------------------------------------------------------------
// 1. Weight repack: Wp[k][eo][ei] = bf16(conv_w[eo][ei][k])
// ---------------------------------------------------------------------------
__global__ __launch_bounds__(256) void pack_w(const float* __restrict__ cw,
                                              unsigned short* __restrict__ Wp) {
    int idx = blockIdx.x * 256 + threadIdx.x;
    if (idx >= KW * E_ * E_) return;
    int k   = idx / (E_ * E_);
    int rem = idx - k * (E_ * E_);
    int eo  = rem >> 9;
    int ei  = rem & (E_ - 1);
    Wp[idx] = f2bf(cw[(eo * E_ + ei) * KW + k]);
}

// ---------------------------------------------------------------------------
// 2. x transpose+cast: Xt[b][l][e] = bf16(x[b][e][l]); rows >= L_ are zero.
// ---------------------------------------------------------------------------
__global__ __launch_bounds__(256) void pack_x(const float* __restrict__ x,
                                              unsigned short* __restrict__ Xt) {
    __shared__ float t[32][33];
    int l0 = blockIdx.x * 32;
    int e0 = blockIdx.y * 32;
    int b  = blockIdx.z;
    int tx = threadIdx.x;        // 0..31
    int ty = threadIdx.y;        // 0..7
    for (int ii = ty; ii < 32; ii += 8) {
        int e = e0 + ii, l = l0 + tx;
        float v = (l < L_) ? x[((size_t)b * E_ + e) * L_ + l] : 0.0f;
        t[ii][tx] = v;
    }
    __syncthreads();
    for (int jj = ty; jj < 32; jj += 8) {
        int l = l0 + jj, e = e0 + tx;
        if (l < XROWS) Xt[((size_t)b * XROWS + l) * E_ + e] = f2bf(t[tx][jj]);
    }
}

// ---------------------------------------------------------------------------
// 3. Conv-as-GEMM with bf16 WMMA + double-buffered async LDS staging.
//    Block: 256 thr = 8 waves; block tile 64(l) x 128(eo).
//    Per K-step (tap,c): A tile 64x32 bf16 (4KB), B tile 128x32 bf16 (8KB)
//    staged by 3 global_load_async_to_lds_b128 per thread.
// ---------------------------------------------------------------------------
__global__ __launch_bounds__(256) void conv_wmma(const unsigned short* __restrict__ Xt,
                                                 const unsigned short* __restrict__ Wp,
                                                 const float* __restrict__ bias,
                                                 float* __restrict__ yT) {
    __shared__ unsigned short smA[2][64 * 32];    //  8 KB total
    __shared__ unsigned short smB[2][128 * 32];   // 16 KB total

    int b    = blockIdx.z;
    int l0   = blockIdx.y * 64;
    int eo0  = blockIdx.x * 128;
    int tid  = threadIdx.x;
    int lane = tid & 31;
    int wave = tid >> 5;
    int subl = wave & 3;                  // which 16-row l strip
    int eoW  = (wave >> 2) * 64;          // eo half: 0 or 64
    int half = (lane >> 4) & 1;           // lane group within wave
    int m    = lane & 15;                 // A: row-in-tile / B,D: column-in-tile

    v8f acc[4] = {v8f{0,0,0,0,0,0,0,0}, v8f{0,0,0,0,0,0,0,0},
                  v8f{0,0,0,0,0,0,0,0}, v8f{0,0,0,0,0,0,0,0}};

    // stage tiles for K-step `step` (tap = step>>4, c-chunk = step&15)
    auto issue = [&](int step, int buf) {
        int tap = step >> 4, c = step & 15;
        const unsigned short* xbase =
            Xt + ((size_t)b * XROWS + l0 + tap) * E_ + c * 32;
        const unsigned short* wbase =
            Wp + ((size_t)tap * E_ + eo0) * E_ + c * 32;
        {   // A tile: 64 rows x 64B = 256 x 16B chunks
            int row = tid >> 2, seg = tid & 3;
            async_copy16((unsigned)(size_t)&smA[buf][row * 32 + seg * 8],
                         xbase + (size_t)row * E_ + seg * 8);
        }
        #pragma unroll
        for (int it = 0; it < 2; ++it) {  // B tile: 128 rows x 64B = 512 chunks
            int id  = it * 256 + tid;
            int row = id >> 2, seg = id & 3;
            async_copy16((unsigned)(size_t)&smB[buf][row * 32 + seg * 8],
                         wbase + (size_t)row * E_ + seg * 8);
        }
    };

    auto compute = [&](int buf) {
        const unsigned short* sa = smA[buf];
        const unsigned short* sb = smB[buf];
        // A fragment (ISA 16-bit A 16x32 layout):
        //  lanes 0-15: K 0..7 & 16..23; lanes 16-31: K 8..15 & 24..31
        Frag A;
        int arow = (subl * 16 + m) * 32;
        A.q[0] = *(const uint4*)&sa[arow + half * 8];
        A.q[1] = *(const uint4*)&sa[arow + 16 + half * 8];
        #pragma unroll
        for (int s = 0; s < 4; ++s) {
            int eoL = eoW + s * 16 + m;   // column N for this lane
            Frag Bf;                       // lanes 0-15: K 0..7 & 16..23; 16-31: K 8..15 & 24..31
            Bf.q[0] = *(const uint4*)&sb[eoL * 32 + half * 8];
            Bf.q[1] = *(const uint4*)&sb[eoL * 32 + 16 + half * 8];
            acc[s] = __builtin_amdgcn_wmma_f32_16x16x32_bf16(
                false, A.v, false, Bf.v, (short)0, acc[s], false, false);
        }
    };

    const int NSTEP = KW * (E_ / 32);     // 80
    issue(0, 0);
    for (int step = 0; step < NSTEP - 1; ++step) {
        issue(step + 1, (step + 1) & 1);
        wait_async<3>();                  // this step's 3 loads landed (in-order)
        __syncthreads();
        compute(step & 1);
        __syncthreads();                  // tile consumed before it is re-filled
    }
    wait_async<0>();
    __syncthreads();
    compute((NSTEP - 1) & 1);

    // D layout: VGPR r holds (M = r + 8*half, N = lane&15).
    #pragma unroll
    for (int s = 0; s < 4; ++s) {
        int eo = eo0 + eoW + s * 16 + m;
        float bv = bias[eo];
        #pragma unroll
        for (int r = 0; r < 8; ++r) {
            int l = l0 + subl * 16 + 8 * half + r;
            if (l < L2_)
                yT[((size_t)b * L_ + l) * E_ + eo] = acc[s][r] + bv;
        }
    }
}

// ---------------------------------------------------------------------------
// 4. Per-position score: s[b][l] = dot(yT[b][l][:], score_w). One wave per row.
// ---------------------------------------------------------------------------
__global__ __launch_bounds__(256) void score_k(const float* __restrict__ yT,
                                               const float* __restrict__ sw,
                                               float* __restrict__ s) {
    int b    = blockIdx.y;
    int wave = threadIdx.x >> 5;
    int lane = threadIdx.x & 31;
    int l    = blockIdx.x * 8 + wave;
    if (l >= L2_) return;
    const float* row = yT + ((size_t)b * L_ + l) * E_;
    float v = 0.0f;
    #pragma unroll 4
    for (int e = lane; e < E_; e += 32) v += row[e] * sw[e];
    #pragma unroll
    for (int off = 16; off > 0; off >>= 1) v += __shfl_down(v, off, 32);
    if (lane == 0) s[(size_t)b * L_ + l] = v;
}

// ---------------------------------------------------------------------------
// 5. Softmax weights A[b][l][4] from window means of s (4092 % w == 0 for w<=4,
//    so all windows are full and in-range).
// ---------------------------------------------------------------------------
__global__ __launch_bounds__(256) void weights_k(const float* __restrict__ s,
                                                 float* __restrict__ A) {
    int idx = blockIdx.x * 256 + threadIdx.x;
    if (idx >= B_ * L2_) return;
    int b = idx / L2_;
    int l = idx - b * L2_;
    const float* sb = s + (size_t)b * L_;
    float sc0 = sb[l];
    int n2 = l & ~1;
    float sc1 = 0.5f * (sb[n2] + sb[n2 + 1]);
    int n3 = (l / 3) * 3;
    float sc2 = (sb[n3] + sb[n3 + 1] + sb[n3 + 2]) * (1.0f / 3.0f);
    int n4 = l & ~3;
    float sc3 = 0.25f * (sb[n4] + sb[n4 + 1] + sb[n4 + 2] + sb[n4 + 3]);
    float mx = fmaxf(fmaxf(sc0, sc1), fmaxf(sc2, sc3));
    float e0 = __expf(sc0 - mx), e1 = __expf(sc1 - mx),
          e2 = __expf(sc2 - mx), e3 = __expf(sc3 - mx);
    float inv = 1.0f / (e0 + e1 + e2 + e3);
    float4 w = make_float4(e0 * inv, e1 * inv, e2 * inv, e3 * inv);
    *(float4*)(A + ((size_t)b * L_ + l) * 4) = w;
}

// ---------------------------------------------------------------------------
// 6. Softmax-weighted multi-scale combine + final stride-2 average pool.
// ---------------------------------------------------------------------------
__global__ __launch_bounds__(256) void combine_k(const float* __restrict__ yT,
                                                 const float* __restrict__ A,
                                                 float* __restrict__ out) {
    __shared__ float  ly[68][65];
    __shared__ float4 lA[64];
    int b   = blockIdx.z;
    int e0  = blockIdx.y * 64;
    int n0  = blockIdx.x * 32;
    int tid = threadIdx.x;
    int lbase = 2 * n0 - 2;

    for (int i = tid; i < 68 * 64; i += 256) {
        int rr = i >> 6, cc = i & 63;
        int l = lbase + rr;
        float v = 0.0f;
        if (l >= 0 && l < L2_)
            v = yT[((size_t)b * L_ + l) * E_ + e0 + cc];
        ly[rr][cc] = v;
    }
    for (int i = tid; i < 64; i += 256) {
        int l = 2 * n0 + i;
        lA[i] = (l < L2_) ? *(const float4*)(A + ((size_t)b * L_ + l) * 4)
                          : make_float4(0, 0, 0, 0);
    }
    __syncthreads();

    int nloc = tid & 31;
    int n    = n0 + nloc;
    int erow = tid >> 5;
    #pragma unroll
    for (int p = 0; p < 8; ++p) {
        int el = erow + p * 8;
        float o = 0.0f;
        if (n < (L2_ / 2)) {
            #pragma unroll
            for (int t = 0; t < 2; ++t) {
                int l  = 2 * n + t;
                float4 w = lA[2 * nloc + t];
                int r1 = l - lbase;
                float p1 = ly[r1][el];
                int r2 = (l & ~1) - lbase;
                float p2 = 0.5f * (ly[r2][el] + ly[r2 + 1][el]);
                int r3 = (l / 3) * 3 - lbase;
                float p3 = (ly[r3][el] + ly[r3 + 1][el] + ly[r3 + 2][el]) *
                           (1.0f / 3.0f);
                int r4 = (l & ~3) - lbase;
                float p4 = 0.25f * (ly[r4][el] + ly[r4 + 1][el] +
                                    ly[r4 + 2][el] + ly[r4 + 3][el]);
                o += w.x * p1 + w.y * p2 + w.z * p3 + w.w * p4;
            }
            o *= 0.5f;
        }
        out[((size_t)b * E_ + (e0 + el)) * ND + n] = o;
    }
}

// ---------------------------------------------------------------------------
// Launch
// ---------------------------------------------------------------------------
extern "C" void kernel_launch(void* const* d_in, const int* in_sizes, int n_in,
                              void* d_out, int out_size, void* d_ws, size_t ws_size,
                              hipStream_t stream) {
    const float* x   = (const float*)d_in[0];   // [B][E][L]
    const float* cw  = (const float*)d_in[1];   // [E][E][K]
    const float* cb  = (const float*)d_in[2];   // [E]
    const float* swv = (const float*)d_in[3];   // [E]
    float* out = (float*)d_out;                 // [B][E][ND]

    char* ws = (char*)d_ws;
    size_t offWp = 0;
    size_t szWp  = (size_t)KW * E_ * E_ * sizeof(unsigned short);      // 2.6 MB
    size_t offXt = (offWp + szWp + 255) & ~(size_t)255;
    size_t szXt  = (size_t)B_ * XROWS * E_ * sizeof(unsigned short);   // 33.6 MB
    size_t offY  = (offXt + szXt + 255) & ~(size_t)255;
    size_t szY   = (size_t)B_ * L_ * E_ * sizeof(float);               // 67.1 MB
    size_t offS  = (offY + szY + 255) & ~(size_t)255;
    size_t szS   = (size_t)B_ * L_ * sizeof(float);                    // 0.13 MB
    size_t offA  = (offS + szS + 255) & ~(size_t)255;

    unsigned short* Wp = (unsigned short*)(ws + offWp);
    unsigned short* Xt = (unsigned short*)(ws + offXt);
    float* yT = (float*)(ws + offY);
    float* sS = (float*)(ws + offS);
    float* aA = (float*)(ws + offA);

    // 1. weight repack
    pack_w<<<dim3((KW * E_ * E_ + 255) / 256), dim3(256), 0, stream>>>(cw, Wp);
    // 2. x transpose (129 l-tiles covers 4096 real + 8 zero-pad rows)
    pack_x<<<dim3(129, E_ / 32, B_), dim3(32, 8), 0, stream>>>(x, Xt);
    // 3. WMMA conv-GEMM: grid (eo tiles, l tiles, b)
    conv_wmma<<<dim3(E_ / 128, L_ / 64, B_), dim3(256), 0, stream>>>(Xt, Wp, cb, yT);
    // 4. scores
    score_k<<<dim3(L_ / 8, B_), dim3(256), 0, stream>>>(yT, swv, sS);
    // 5. softmax weights
    weights_k<<<dim3((B_ * L2_ + 255) / 256), dim3(256), 0, stream>>>(sS, aA);
    // 6. combine + final pool
    combine_k<<<dim3(ND / 32, E_ / 64, B_), dim3(256), 0, stream>>>(yT, aA, out);
}
